// LinearAttentionLayer_37589553774701
// MI455X (gfx1250) — matrix operand order repaired and checked by reading, not compiled, not run on hardware
//
#include <hip/hip_runtime.h>

// ---------------------------------------------------------------------------
// LinearAttentionLayer for MI455X (gfx1250, wave32, WMMA + async LDS copy)
// B=4, S=4096, H=1024, A=1024
//
// Pipeline (per batch, looped on stream; workspace ~63 MB):
//   1) fp32 -> bf16 convert (weights once, hidden slab per batch)
//   2) fused WMMA GEMM with double-buffered async global->LDS tile staging:
//      q = hid@wq^T, k = hid@wk^T, v = hid@wv^T
//      (v_wmma_f32_16x16x32_bf16, fp32 accumulate), stores q, k, u=k*v (fp32)
//   3) segmented scan over S (8 segments): seg-sums -> exclusive carry ->
//      carry-initialized local scans + shuffle-reduced dot partials
//      (fp32, fully deterministic -> no float atomics)
//   4) reduce partials -> scores[s] = num/(den+1e-6)
//   5) out[b,s,h] = scores[s] * rowsum(wo)[h]   (float4 stores)
// ---------------------------------------------------------------------------

typedef __attribute__((ext_vector_type(16))) __bf16 v16bf;
typedef __attribute__((ext_vector_type(8)))  __bf16 v8bf;
typedef __attribute__((ext_vector_type(4)))  __bf16 v4bf;
typedef __attribute__((ext_vector_type(8)))  float  v8f;
typedef __attribute__((ext_vector_type(4)))  float  v4f;
typedef __attribute__((ext_vector_type(4)))  int    v4i;

// address-space-qualified int4 for the async-copy builtin
// (param 0 is 'int4 __device__ *' = AS1, param 1 is the LDS side = AS3)
typedef __attribute__((address_space(1))) v4i as1_v4i;
typedef __attribute__((address_space(3))) v4i as3_v4i;

#define DIM_H 1024
#define DIM_A 1024
#define SEQ_S 4096
#define NBATCH 4
#define NSEG   8
#define SEGLEN (SEQ_S / NSEG)     // 512
#define NCHUNK 32                 // 32 a-lanes per scan wave

// ---- gfx1250 async global->LDS path (guarded; fallback = VGPR bounce) -----
#if defined(__has_builtin)
#if __has_builtin(__builtin_amdgcn_global_load_async_to_lds_b128) && \
    __has_builtin(__builtin_amdgcn_s_wait_asynccnt)
#define LA_ASYNC_LDS 1
#endif
#endif

__device__ __forceinline__ void la_cp16(const __bf16* __restrict__ g,
                                        char* __restrict__ l) {
#ifdef LA_ASYNC_LDS
    __builtin_amdgcn_global_load_async_to_lds_b128(
        (as1_v4i*)g, (as3_v4i*)l, 0, 0);
#else
    *(v4i*)l = *(const v4i*)g;
#endif
}

__device__ __forceinline__ void la_async_drain() {
#ifdef LA_ASYNC_LDS
    __builtin_amdgcn_s_wait_asynccnt(0);
#endif
}

// -------------------------------------------------- fp32 -> bf16 (4-wide) --
__global__ void la_cvt_f32_bf16_x4(const float4* __restrict__ in,
                                   v4bf* __restrict__ out, int n4) {
    int i = blockIdx.x * blockDim.x + threadIdx.x;
    int stride = gridDim.x * blockDim.x;
    for (; i < n4; i += stride) {
        float4 x = in[i];
        v4bf y;
        y[0] = (__bf16)x.x; y[1] = (__bf16)x.y;
        y[2] = (__bf16)x.z; y[3] = (__bf16)x.w;
        out[i] = y;
    }
}

// -------------------------------------------------- rowsum(wo) over A -----
__global__ void la_wo_rowsum(const float* __restrict__ wo,
                             float* __restrict__ rs) {
    __shared__ float sm[256];
    const int h = blockIdx.x;
    float acc = 0.f;
    for (int a = threadIdx.x; a < DIM_A; a += 256)
        acc += wo[(size_t)h * DIM_A + a];
    sm[threadIdx.x] = acc;
    __syncthreads();
    for (int off = 128; off > 0; off >>= 1) {
        if ((int)threadIdx.x < off) sm[threadIdx.x] += sm[threadIdx.x + off];
        __syncthreads();
    }
    if (threadIdx.x == 0) rs[h] = sm[0];
}

// -------------------------------------------------- WMMA fragment loaders --
// LDS tiles: 64 rows x 32 bf16, row pitch 80 B (64 B payload + 16 B pad so
// 16-lane ds_load_b128 bursts land on distinct bank groups).
#define TILE_PITCH 80
#define TILE_BYTES (64 * TILE_PITCH)          // 5120 B
#define STAGE_BYTES (4 * TILE_BYTES)          // A + Wq + Wk + Wv = 20480 B

// A fragment (16x32 bf16, ISA 7.12.2): lane L row M=L&15; elements j=0..7 ->
// K = j + 8*hi, j=8..15 -> K = 16 + 8*hi + (j-8): two 16B chunks 32B apart.
__device__ __forceinline__ v16bf la_frag_a(const char* __restrict__ tile,
                                           int rowLocal, int laneHi) {
    const char* p = tile + rowLocal * TILE_PITCH + laneHi * 16;
    v8bf lo = *(const v8bf*)p;
    v8bf hi = *(const v8bf*)(p + 32);
    v16bf r;
#pragma unroll
    for (int j = 0; j < 8; ++j) { r[j] = lo[j]; r[j + 8] = hi[j]; }
    return r;
}

// B fragment (32x16 bf16): lane L column N=L&15; elements j=0..7 ->
// K = j + 8*hi, j=8..15 -> K = 16 + 8*hi + (j-8): two 16B chunks 32B apart.
__device__ __forceinline__ v16bf la_frag_b(const char* __restrict__ tile,
                                           int rowLocal, int laneHi) {
    const char* p = tile + rowLocal * TILE_PITCH + laneHi * 16;
    v8bf lo = *(const v8bf*)p;
    v8bf hi = *(const v8bf*)(p + 32);
    v16bf r;
#pragma unroll
    for (int j = 0; j < 8; ++j) { r[j] = lo[j]; r[j + 8] = hi[j]; }
    return r;
}

// Cooperative stage of one K-step (A rows srow0..+63, W rows col0..+63,
// K = h0..h0+31) into an LDS buffer. 128 threads x 8 x 16B = 16 KB payload.
__device__ __forceinline__ void la_stage(const __bf16* __restrict__ hid,
                                         const __bf16* __restrict__ wq,
                                         const __bf16* __restrict__ wk,
                                         const __bf16* __restrict__ wv,
                                         int srow0, int col0, int h0,
                                         char* __restrict__ buf, int tid) {
#pragma unroll
    for (int it = 0; it < 2; ++it) {
        const int c   = tid * 2 + it;      // 0..255 : 64 rows x 4 chunks
        const int row = c >> 2;
        const int sub = c & 3;
        const int ldo = row * TILE_PITCH + sub * 16;
        const size_t ga = (size_t)(srow0 + row) * DIM_H + h0 + sub * 8;
        const size_t gw = (size_t)(col0 + row) * DIM_H + h0 + sub * 8;
        la_cp16(hid + ga, buf + ldo);
        la_cp16(wq  + gw, buf + TILE_BYTES     + ldo);
        la_cp16(wk  + gw, buf + 2 * TILE_BYTES + ldo);
        la_cp16(wv  + gw, buf + 3 * TILE_BYTES + ldo);
    }
}

// -------------------------------------------------- fused q/k/u GEMM -------
// block = 128 threads (4 waves). Block tile 64(M=s) x 64(N=a); each wave owns
// 16(M) x 64(N) and accumulates q,k,v together. All operands come from LDS
// (double-buffered async staging), so the 12 WMMAs per K-step are not gated
// on global-load latency.
__global__ __launch_bounds__(128, 1)
void la_qku_gemm_wmma(const __bf16* __restrict__ hid,   // [S,H] bf16
                      const __bf16* __restrict__ wqb,   // [A,H] bf16
                      const __bf16* __restrict__ wkb,
                      const __bf16* __restrict__ wvb,
                      float* __restrict__ qo,           // [S,A] f32
                      float* __restrict__ ko,
                      float* __restrict__ uo) {         // u = k*v
    __shared__ char smem[2 * STAGE_BYTES];              // 40 KB, double buffer

    const int tid    = threadIdx.x;
    const int lane   = tid & 31;
    const int wave   = tid >> 5;
    const int laneHi = (lane >> 4) & 1;
    const int srow0  = blockIdx.x * 64;                 // block s base
    const int col0   = blockIdx.y * 64;                 // block a base
    const int row0   = srow0 + wave * 16;               // wave s base

    v8f accq[4] = {{0.f}}; v8f acck[4] = {{0.f}}; v8f accv[4] = {{0.f}};

    la_stage(hid, wqb, wkb, wvb, srow0, col0, 0, smem, tid);

    constexpr int KSTEPS = DIM_H / 32;
    for (int i = 0; i < KSTEPS; ++i) {
        char* buf = smem + (size_t)(i & 1) * STAGE_BYTES;

        la_async_drain();        // our stage of buf (i&1) has landed
        __syncthreads();         // everyone's stage landed; everyone done
                                 // reading buf ((i+1)&1) from step i-1

        if (i + 1 < KSTEPS)
            la_stage(hid, wqb, wkb, wvb, srow0, col0, (i + 1) * 32,
                     smem + (size_t)((i + 1) & 1) * STAGE_BYTES, tid);

        const v16bf af = la_frag_a(buf, wave * 16 + (lane & 15), laneHi);
#pragma unroll
        for (int nt = 0; nt < 4; ++nt) {
            const int brow = nt * 16 + (lane & 15);
            v16bf bq = la_frag_b(buf + TILE_BYTES,     brow, laneHi);
            v16bf bk = la_frag_b(buf + 2 * TILE_BYTES, brow, laneHi);
            v16bf bv = la_frag_b(buf + 3 * TILE_BYTES, brow, laneHi);
            accq[nt] = __builtin_amdgcn_wmma_f32_16x16x32_bf16(
                false, af, false, bq, (short)0, accq[nt], false, false);
            acck[nt] = __builtin_amdgcn_wmma_f32_16x16x32_bf16(
                false, af, false, bk, (short)0, acck[nt], false, false);
            accv[nt] = __builtin_amdgcn_wmma_f32_16x16x32_bf16(
                false, af, false, bv, (short)0, accv[nt], false, false);
        }
    }

    // C/D layout: VGPR r -> M = r + 8*(lane>=16), N = lane&15
#pragma unroll
    for (int nt = 0; nt < 4; ++nt) {
        const int acol = col0 + nt * 16 + (lane & 15);
#pragma unroll
        for (int r = 0; r < 8; ++r) {
            const int srow = row0 + r + laneHi * 8;
            const size_t o = (size_t)srow * DIM_A + acol;
            const float kk = acck[nt][r];
            qo[o] = accq[nt][r];
            ko[o] = kk;
            uo[o] = kk * accv[nt][r];
        }
    }
}

// -------------------------------------------------- segmented causal scan --
// Phase A: per-(chunk,segment) sums of k and u over the 512-step segment.
__global__ __launch_bounds__(32)
void la_seg_sums(const float* __restrict__ kb, const float* __restrict__ ub,
                 float* __restrict__ segk, float* __restrict__ segu) {
    const int a   = blockIdx.x * 32 + threadIdx.x;
    const int seg = blockIdx.y;
    float sk = 0.f, su = 0.f;
    for (int s = seg * SEGLEN; s < (seg + 1) * SEGLEN; ++s) {
        const size_t idx = (size_t)s * DIM_A + a;
        sk += kb[idx];
        su += ub[idx];
    }
    segk[seg * DIM_A + a] = sk;
    segu[seg * DIM_A + a] = su;
}

// Phase B: in-place exclusive prefix over the 8 segments (per a).
__global__ void la_seg_scan(float* __restrict__ segk,
                            float* __restrict__ segu) {
    const int a = blockIdx.x * blockDim.x + threadIdx.x;
    if (a >= DIM_A) return;
    float rk = 0.f, ru = 0.f;
#pragma unroll
    for (int seg = 0; seg < NSEG; ++seg) {
        const float tk = segk[seg * DIM_A + a];
        const float tu = segu[seg * DIM_A + a];
        segk[seg * DIM_A + a] = rk;
        segu[seg * DIM_A + a] = ru;
        rk += tk;
        ru += tu;
    }
}

// Phase C: carry-initialized local scan + shuffle-reduced dot partials.
__global__ __launch_bounds__(32)
void la_scan_partial(const float* __restrict__ qb,
                     const float* __restrict__ kb,
                     const float* __restrict__ ub,
                     const float* __restrict__ segk,
                     const float* __restrict__ segu,
                     float* __restrict__ pnum,    // [NCHUNK][S]
                     float* __restrict__ pden) {
    const int lane  = threadIdx.x;
    const int chunk = blockIdx.x;    // 0..31
    const int seg   = blockIdx.y;    // 0..7
    const int a = chunk * 32 + lane;
    float kc = segk[seg * DIM_A + a];
    float uc = segu[seg * DIM_A + a];
    for (int s = seg * SEGLEN; s < (seg + 1) * SEGLEN; ++s) {
        const size_t idx = (size_t)s * DIM_A + a;
        const float q = qb[idx];
        kc += kb[idx];
        uc += ub[idx];
        float pn = q * uc;
        float pd = q * kc;
#pragma unroll
        for (int off = 16; off > 0; off >>= 1) {
            pn += __shfl_xor(pn, off, 32);
            pd += __shfl_xor(pd, off, 32);
        }
        if (lane == 0) {
            pnum[chunk * SEQ_S + s] = pn;
            pden[chunk * SEQ_S + s] = pd;
        }
    }
}

__global__ void la_reduce_scores(const float* __restrict__ pnum,
                                 const float* __restrict__ pden,
                                 float* __restrict__ scores) {
    const int s = blockIdx.x * blockDim.x + threadIdx.x;
    if (s >= SEQ_S) return;
    float n = 0.f, d = 0.f;
#pragma unroll 8
    for (int c = 0; c < NCHUNK; ++c) {
        n += pnum[c * SEQ_S + s];
        d += pden[c * SEQ_S + s];
    }
    scores[s] = n / (d + 1e-6f);
}

// -------------------------------------------------- broadcast output -------
__global__ void la_finalize(const float* __restrict__ scores,
                            const float* __restrict__ rs,
                            float* __restrict__ out) {
    const int t = blockIdx.x * blockDim.x + threadIdx.x;   // over S*H/4
    const int s  = t / (DIM_H / 4);
    const int h4 = t % (DIM_H / 4);
    const float sc = scores[s];
    v4f w = ((const v4f*)rs)[h4];
    ((v4f*)out)[t] = sc * w;
}

// ---------------------------------------------------------------------------
extern "C" void kernel_launch(void* const* d_in, const int* in_sizes, int n_in,
                              void* d_out, int out_size, void* d_ws, size_t ws_size,
                              hipStream_t stream) {
    (void)in_sizes; (void)n_in; (void)out_size; (void)ws_size;
    const float* hidden = (const float*)d_in[0];   // [B,S,H]
    const float* wq     = (const float*)d_in[1];   // [A,H]
    const float* wk     = (const float*)d_in[2];
    const float* wv     = (const float*)d_in[3];
    const float* wo     = (const float*)d_in[4];   // [H,A]
    float* out = (float*)d_out;                    // [B,S,H]

    // ---- workspace carve-up (~63 MB) ----
    char* base = (char*)d_ws;
    size_t off = 0;
    auto carve = [&](size_t bytes) -> char* {
        char* p = base + off;
        off = (off + bytes + 255) & ~(size_t)255;
        return p;
    };
    __bf16* wqb   = (__bf16*)carve((size_t)DIM_A * DIM_H * 2);
    __bf16* wkb   = (__bf16*)carve((size_t)DIM_A * DIM_H * 2);
    __bf16* wvb   = (__bf16*)carve((size_t)DIM_A * DIM_H * 2);
    float*  rs    = (float*) carve((size_t)DIM_H * 4);
    __bf16* hidbf = (__bf16*)carve((size_t)SEQ_S * DIM_H * 2);
    float*  qbuf  = (float*) carve((size_t)SEQ_S * DIM_A * 4);
    float*  kbuf  = (float*) carve((size_t)SEQ_S * DIM_A * 4);
    float*  ubuf  = (float*) carve((size_t)SEQ_S * DIM_A * 4);
    float*  pnum  = (float*) carve((size_t)NCHUNK * SEQ_S * 4);
    float*  pden  = (float*) carve((size_t)NCHUNK * SEQ_S * 4);
    float*  segk  = (float*) carve((size_t)NSEG * DIM_A * 4);
    float*  segu  = (float*) carve((size_t)NSEG * DIM_A * 4);
    float*  scr   = (float*) carve((size_t)SEQ_S * 4);

    // ---- weights: convert once, rowsum(wo) once ----
    {
        const int n4 = (DIM_A * DIM_H) / 4;
        la_cvt_f32_bf16_x4<<<1024, 256, 0, stream>>>((const float4*)wq, (v4bf*)wqb, n4);
        la_cvt_f32_bf16_x4<<<1024, 256, 0, stream>>>((const float4*)wk, (v4bf*)wkb, n4);
        la_cvt_f32_bf16_x4<<<1024, 256, 0, stream>>>((const float4*)wv, (v4bf*)wvb, n4);
        la_wo_rowsum<<<DIM_H, 256, 0, stream>>>(wo, rs);
    }

    // ---- per-batch pipeline ----
    for (int b = 0; b < NBATCH; ++b) {
        const float* hslab = hidden + (size_t)b * SEQ_S * DIM_H;
        la_cvt_f32_bf16_x4<<<2048, 256, 0, stream>>>(
            (const float4*)hslab, (v4bf*)hidbf, (SEQ_S * DIM_H) / 4);

        dim3 ggrid(SEQ_S / 64, DIM_A / 64);     // 64 x 16 blocks
        la_qku_gemm_wmma<<<ggrid, 128, 0, stream>>>(
            hidbf, wqb, wkb, wvb, qbuf, kbuf, ubuf);

        dim3 sgrid(NCHUNK, NSEG);               // 32 x 8 blocks
        la_seg_sums<<<sgrid, 32, 0, stream>>>(kbuf, ubuf, segk, segu);
        la_seg_scan<<<DIM_A / 256, 256, 0, stream>>>(segk, segu);
        la_scan_partial<<<sgrid, 32, 0, stream>>>(
            qbuf, kbuf, ubuf, segk, segu, pnum, pden);

        la_reduce_scores<<<SEQ_S / 256, 256, 0, stream>>>(pnum, pden, scr);

        la_finalize<<<(SEQ_S * DIM_H / 4) / 256, 256, 0, stream>>>(
            scr, rs, out + (size_t)b * SEQ_S * DIM_H);
    }
}
